// CrossAttention_5265629905601
// MI455X (gfx1250) — compile-verified, matrix operand order checked by reading, not run on hardware
//
#include <hip/hip_runtime.h>
#include <cstdint>

// CrossAttention for MI455X (gfx1250, wave32, WMMA + async-to-LDS).
// Pipeline: 3x projection GEMM (f32->f16, V transposed) -> scores GEMM +
// in-block softmax (f16 scores rewritten in place as probs) -> P@V GEMM with
// 1/rowsum epilogue (fp32 out).
// Workspace requirement: ~336 MB (q,k f16 64MB each; v^T f16 64MB;
// scores/probs f16 128MB in place; rowsum 128KB).

typedef __attribute__((ext_vector_type(16))) _Float16 v16h;
typedef __attribute__((ext_vector_type(8)))  _Float16 v8h;
typedef __attribute__((ext_vector_type(8)))  float    v8f;

#define LQ   2048
#define NBAT 16
#define ODIM 1024

union V16H { v16h v; v8h h[2]; };

// Load one 16x32 f16 WMMA operand fragment from a row-major tile (ld = 32).
// Per ISA layout: lane = 16*half + r; VGPRs 0..3 hold K = 0..7 (+8*half),
// VGPRs 4..7 hold K = 16..23 (+8*half) -> two contiguous 16B reads per lane.
__device__ __forceinline__ v16h frag16(const _Float16* p) {
  V16H f;
  f.h[0] = *(const v8h*)(p);
  f.h[1] = *(const v8h*)(p + 16);
  return f.v;
}

__device__ __forceinline__ v8f wmma16(v16h a, v16h b, v8f c) {
  // (neg_a, A, neg_b, B, c_mod, C, reuse_a, reuse_b)
  return __builtin_amdgcn_wmma_f32_16x16x32_f16(false, a, false, b,
                                                (short)0, c, false, false);
}

// --- CDNA5 async global->LDS copy (ASYNCcnt-tracked), GV addressing mode ---
// Flat LDS aperture keeps the LDS byte offset in addr[31:0] (ISA 10.2).
__device__ __forceinline__ uint32_t lds_addr32(const void* p) {
  return (uint32_t)(uintptr_t)p;
}
__device__ __forceinline__ void async_ld16(uint32_t lds, const void* g) {
  asm volatile("global_load_async_to_lds_b128 %0, %1, off"
               :: "v"(lds), "v"(g) : "memory");
}
__device__ __forceinline__ void wait_async4() {
  asm volatile("s_wait_asynccnt 0x4" ::: "memory");
}
__device__ __forceinline__ void wait_async0() {
  asm volatile("s_wait_asynccnt 0x0" ::: "memory");
}

// Stage a 128x32 f16 tile (row-major, leading dim ldg) into LDS via async
// copies: 512 x 16B chunks, 2 per thread (=> 2 async ops/lane, 2/wave-slot).
__device__ __forceinline__ void stage_async_128x32(uint32_t ldsBase,
                                                   const _Float16* gBase,
                                                   size_t ldg, int ks, int tid) {
  #pragma unroll
  for (int q = 0; q < 2; ++q) {
    const int g   = tid * 2 + q;     // 512 groups
    const int row = g >> 2;
    const int c8  = (g & 3) * 8;
    async_ld16(ldsBase + (uint32_t)(row * 32 + c8) * 2,
               (const void*)(gBase + (size_t)row * ldg + ks + c8));
  }
}

// ---------------------------------------------------------------------------
// Kernel 1: projection GEMM  Y[m][n] = sum_k X[m][k]*W[n][k] + bias[n]
// X: [32768][K] f32, W: [1024][K] f32, Y: f16 ([m][1024], or transposed
// [b][1024][2048] when transposedV != 0). Needs f32->f16 conversion, so it
// stages through VGPRs (v_cvt_pk_f16_f32 + ds_store_b128).
// Block tile 128x64, 8 waves in 4x2, each wave a 2x2 grid of 16x16 WMMA tiles.
// ---------------------------------------------------------------------------
__global__ __launch_bounds__(256)
void proj_kernel(const float* __restrict__ X, const float* __restrict__ W,
                 const float* __restrict__ bias, _Float16* __restrict__ Y,
                 int K, int transposedV)
{
  __shared__ __align__(16) _Float16 sA[128 * 32];
  __shared__ __align__(16) _Float16 sB[64 * 32];

  const int tid  = threadIdx.x;
  const int lane = tid & 31;
  const int wid  = tid >> 5;
  const int l15  = lane & 15;
  const int hf   = lane >> 4;
  const int wm   = wid & 3;     // row group of 32
  const int wn   = wid >> 2;    // col group of 32
  const int m0   = blockIdx.x * 128;
  const int n0   = blockIdx.y * 64;

  v8f acc[2][2];
  #pragma unroll
  for (int i = 0; i < 2; ++i)
    #pragma unroll
    for (int j = 0; j < 2; ++j)
      acc[i][j] = {};

  for (int ks = 0; ks < K; ks += 32) {
    __syncthreads();
    // Stage A: 128x32 (f32 -> f16), 4 float4 per thread
    #pragma unroll
    for (int q = 0; q < 4; ++q) {
      int g   = tid * 4 + q;              // 1024 float4 groups
      int row = g >> 3;
      int c4  = (g & 7) * 4;
      const float4 x = *(const float4*)(X + (size_t)(m0 + row) * K + ks + c4);
      _Float16* d = sA + row * 32 + c4;
      d[0] = (_Float16)x.x; d[1] = (_Float16)x.y;
      d[2] = (_Float16)x.z; d[3] = (_Float16)x.w;
    }
    // Stage B: 64x32 (f32 -> f16), 2 float4 per thread
    #pragma unroll
    for (int q = 0; q < 2; ++q) {
      int g   = tid * 2 + q;              // 512 float4 groups
      int row = g >> 3;
      int c4  = (g & 7) * 4;
      const float4 x = *(const float4*)(W + (size_t)(n0 + row) * K + ks + c4);
      _Float16* d = sB + row * 32 + c4;
      d[0] = (_Float16)x.x; d[1] = (_Float16)x.y;
      d[2] = (_Float16)x.z; d[3] = (_Float16)x.w;
    }
    __syncthreads();

    const _Float16* aB = sA + (wm * 32 + l15) * 32 + hf * 8;
    const _Float16* bB = sB + (wn * 32 + l15) * 32 + hf * 8;
    v16h a0 = frag16(aB);
    v16h a1 = frag16(aB + 16 * 32);
    v16h b0 = frag16(bB);
    v16h b1 = frag16(bB + 16 * 32);
    acc[0][0] = wmma16(a0, b0, acc[0][0]);
    acc[0][1] = wmma16(a0, b1, acc[0][1]);
    acc[1][0] = wmma16(a1, b0, acc[1][0]);
    acc[1][1] = wmma16(a1, b1, acc[1][1]);
  }

  // Epilogue: bias add + f16 store (C layout: element e -> row e + 8*half,
  // col = lane%16).
  #pragma unroll
  for (int j = 0; j < 2; ++j) {
    const int col = n0 + wn * 32 + j * 16 + l15;
    const float bv = bias[col];
    #pragma unroll
    for (int i = 0; i < 2; ++i) {
      const int rbase = wm * 32 + i * 16 + hf * 8;
      if (!transposedV) {
        #pragma unroll
        for (int e = 0; e < 8; ++e) {
          const int row = m0 + rbase + e;
          Y[(size_t)row * ODIM + col] = (_Float16)(acc[i][j][e] + bv);
        }
      } else {
        // V^T layout [b][o][j]: per-lane 8 consecutive rows -> one 16B store
        const int b    = m0 >> 11;          // 2048 rows per batch
        const int mloc = (m0 & 2047) + rbase;
        v8h hv;
        #pragma unroll
        for (int e = 0; e < 8; ++e) hv[e] = (_Float16)(acc[i][j][e] + bv);
        *(v8h*)(Y + (size_t)b * ODIM * LQ + (size_t)col * LQ + mloc) = hv;
      }
    }
  }
}

// ---------------------------------------------------------------------------
// Kernel 2: scores + softmax. Block owns 128 q-rows x all 2048 kv cols of one
// batch. Phase 1: S = q.k^T (raw, f16 to ws) with running row-max in regs,
// 128x128 block tile, double-buffered async-to-LDS staging.
// Phase 2: rewrite in place as exp(scale*(s-max)), store row sums.
// ---------------------------------------------------------------------------
__global__ __launch_bounds__(256)
void attn_softmax_kernel(const _Float16* __restrict__ qb,
                         const _Float16* __restrict__ kb,
                         _Float16* __restrict__ SP,
                         float* __restrict__ rowsum)
{
  __shared__ __align__(16) _Float16 sA[2][128 * 32];
  __shared__ __align__(16) _Float16 sB[2][128 * 32];
  __shared__ float rowmax2[128][2];
  __shared__ float sum2[128][2];

  const int tid  = threadIdx.x;
  const int lane = tid & 31;
  const int wid  = tid >> 5;
  const int l15  = lane & 15;
  const int hf   = lane >> 4;
  const int wm   = wid & 3;     // row group of 32 (of 128)
  const int wn   = wid >> 2;    // col group of 64 (of 128)
  const int b    = blockIdx.y;
  const int q0   = blockIdx.x * 128;

  const _Float16* Aq  = qb + ((size_t)b * LQ + q0) * ODIM;
  _Float16*       SPb = SP + (size_t)b * LQ * LQ;

  const uint32_t aLds[2] = { lds_addr32(&sA[0][0]), lds_addr32(&sA[1][0]) };
  const uint32_t bLds[2] = { lds_addr32(&sB[0][0]), lds_addr32(&sB[1][0]) };

  v8f mx[2];
  #pragma unroll
  for (int i = 0; i < 2; ++i)
    #pragma unroll
    for (int e = 0; e < 8; ++e) mx[i][e] = -3.0e38f;

  for (int chunk = 0; chunk < LQ / 128; ++chunk) {
    const int n0 = chunk * 128;
    const _Float16* Bk = kb + ((size_t)b * LQ + n0) * ODIM;

    v8f acc[2][4];
    #pragma unroll
    for (int i = 0; i < 2; ++i)
      #pragma unroll
      for (int j = 0; j < 4; ++j)
        acc[i][j] = {};

    const int NS = ODIM / 32;   // 32 K-slices
    for (int i = 0; i < NS; ++i) {
      const int p = i & 1;
      __syncthreads();          // all waves done reading buf[1-p] (prev iter)
      if (i == 0) {
        stage_async_128x32(aLds[0], Aq, ODIM, 0, tid);
        stage_async_128x32(bLds[0], Bk, ODIM, 0, tid);
      }
      if (i + 1 < NS) {         // prefetch next slice into the other buffer
        stage_async_128x32(aLds[1 - p], Aq, ODIM, (i + 1) * 32, tid);
        stage_async_128x32(bLds[1 - p], Bk, ODIM, (i + 1) * 32, tid);
        wait_async4();          // 4 ops/slice/wave, in-order: slice i landed
      } else {
        wait_async0();
      }
      __syncthreads();          // everyone's slice-i copies visible

      const _Float16* aB = &sA[p][(wm * 32 + l15) * 32 + hf * 8];
      const _Float16* bB = &sB[p][(wn * 64 + l15) * 32 + hf * 8];
      v16h a0 = frag16(aB);
      v16h a1 = frag16(aB + 16 * 32);
      #pragma unroll
      for (int j = 0; j < 4; ++j) {
        v16h bj = frag16(bB + j * 16 * 32);
        acc[0][j] = wmma16(a0, bj, acc[0][j]);
        acc[1][j] = wmma16(a1, bj, acc[1][j]);
      }
    }

    // Store raw f16 scores + track per-row max (f32) in registers.
    #pragma unroll
    for (int j = 0; j < 4; ++j) {
      const int col = n0 + wn * 64 + j * 16 + l15;
      #pragma unroll
      for (int i = 0; i < 2; ++i) {
        const int rbase = wm * 32 + i * 16 + hf * 8;
        #pragma unroll
        for (int e = 0; e < 8; ++e) {
          const float sval = acc[i][j][e];
          SPb[(size_t)(q0 + rbase + e) * LQ + col] = (_Float16)sval;
          mx[i][e] = fmaxf(mx[i][e], sval);
        }
      }
    }
  }

  // Cross-lane row-max reduction: cols live in lanes 0..15 of each half; the
  // two halves hold different rows, so reduce with masks 1,2,4,8 only.
  #pragma unroll
  for (int mk = 1; mk < 16; mk <<= 1)
    #pragma unroll
    for (int i = 0; i < 2; ++i)
      #pragma unroll
      for (int e = 0; e < 8; ++e)
        mx[i][e] = fmaxf(mx[i][e], __shfl_xor(mx[i][e], mk, 32));

  if (l15 == 0) {
    #pragma unroll
    for (int i = 0; i < 2; ++i)
      #pragma unroll
      for (int e = 0; e < 8; ++e)
        rowmax2[wm * 32 + i * 16 + hf * 8 + e][wn] = mx[i][e];
  }
  __threadfence_block();   // make phase-1 global stores visible in-block
  __syncthreads();

  // Phase 2: in-place exp + row sums. 2 threads per row, 1024 cols each.
  const float scale = 0.03125f;   // 1024^-0.5
  const int row  = tid >> 1;
  const int half = tid & 1;
  const float m = fmaxf(rowmax2[row][0], rowmax2[row][1]);
  float s = 0.0f;
  _Float16* p = SPb + (size_t)(q0 + row) * LQ + half * 1024;
  for (int i0 = 0; i0 < 1024; i0 += 8) {
    v8h x = *(v8h*)(p + i0);
    v8h y;
    #pragma unroll
    for (int e = 0; e < 8; ++e) {
      const float f = __expf(scale * ((float)x[e] - m));
      s += f;
      y[e] = (_Float16)f;
    }
    *(v8h*)(p + i0) = y;
  }
  sum2[row][half] = s;
  __syncthreads();
  if (tid < 128)
    rowsum[(size_t)b * LQ + q0 + tid] = sum2[tid][0] + sum2[tid][1];
}

// ---------------------------------------------------------------------------
// Kernel 3: out = (P @ V) * (1/rowsum), fp32 out. 128x128 block tile,
// double-buffered async-to-LDS staging.
// A = probs f16 [b][2048][2048]; B = V^T f16 [b][1024][2048] (K-major).
// ---------------------------------------------------------------------------
__global__ __launch_bounds__(256)
void attn_pv_kernel(const _Float16* __restrict__ SP,
                    const _Float16* __restrict__ vbt,
                    const float* __restrict__ rowsum,
                    float* __restrict__ out)
{
  __shared__ __align__(16) _Float16 sA[2][128 * 32];
  __shared__ __align__(16) _Float16 sB[2][128 * 32];
  __shared__ float sInv[128];

  const int tid  = threadIdx.x;
  const int lane = tid & 31;
  const int wid  = tid >> 5;
  const int l15  = lane & 15;
  const int hf   = lane >> 4;
  const int wm   = wid & 3;
  const int wn   = wid >> 2;
  const int b    = blockIdx.z;
  const int m0   = blockIdx.x * 128;
  const int o0   = blockIdx.y * 128;

  const _Float16* Ap = SP  + (size_t)b * LQ * LQ + (size_t)m0 * LQ;
  const _Float16* Bv = vbt + (size_t)b * ODIM * LQ + (size_t)o0 * LQ;

  const uint32_t aLds[2] = { lds_addr32(&sA[0][0]), lds_addr32(&sA[1][0]) };
  const uint32_t bLds[2] = { lds_addr32(&sB[0][0]), lds_addr32(&sB[1][0]) };

  if (tid < 128) sInv[tid] = 1.0f / rowsum[(size_t)b * LQ + m0 + tid];

  v8f acc[2][4];
  #pragma unroll
  for (int i = 0; i < 2; ++i)
    #pragma unroll
    for (int j = 0; j < 4; ++j)
      acc[i][j] = {};

  const int NS = LQ / 32;   // 64 K-slices
  for (int i = 0; i < NS; ++i) {
    const int p = i & 1;
    __syncthreads();
    if (i == 0) {
      stage_async_128x32(aLds[0], Ap, LQ, 0, tid);
      stage_async_128x32(bLds[0], Bv, LQ, 0, tid);
    }
    if (i + 1 < NS) {
      stage_async_128x32(aLds[1 - p], Ap, LQ, (i + 1) * 32, tid);
      stage_async_128x32(bLds[1 - p], Bv, LQ, (i + 1) * 32, tid);
      wait_async4();
    } else {
      wait_async0();
    }
    __syncthreads();

    const _Float16* aB = &sA[p][(wm * 32 + l15) * 32 + hf * 8];
    const _Float16* bB = &sB[p][(wn * 64 + l15) * 32 + hf * 8];
    v16h a0 = frag16(aB);
    v16h a1 = frag16(aB + 16 * 32);
    #pragma unroll
    for (int j = 0; j < 4; ++j) {
      v16h bj = frag16(bB + j * 16 * 32);
      acc[0][j] = wmma16(a0, bj, acc[0][j]);
      acc[1][j] = wmma16(a1, bj, acc[1][j]);
    }
  }
  __syncthreads();

  #pragma unroll
  for (int j = 0; j < 4; ++j) {
    const int col = o0 + wn * 64 + j * 16 + l15;
    #pragma unroll
    for (int i = 0; i < 2; ++i) {
      const int rbase = wm * 32 + i * 16 + hf * 8;
      #pragma unroll
      for (int e = 0; e < 8; ++e) {
        const int row = rbase + e;
        out[(size_t)b * LQ * ODIM + (size_t)(m0 + row) * ODIM + col] =
            acc[i][j][e] * sInv[row];
      }
    }
  }
}

// ---------------------------------------------------------------------------
extern "C" void kernel_launch(void* const* d_in, const int* in_sizes, int n_in,
                              void* d_out, int out_size, void* d_ws, size_t ws_size,
                              hipStream_t stream) {
  const float* query = (const float*)d_in[0];   // [16,2048,1024]
  const float* key   = (const float*)d_in[1];   // [16,2048,768]
  const float* value = (const float*)d_in[2];   // [16,2048,768]
  const float* Wq    = (const float*)d_in[3];   // [1024,1024]
  const float* bq    = (const float*)d_in[4];
  const float* Wk    = (const float*)d_in[5];   // [1024,768]
  const float* bk    = (const float*)d_in[6];
  const float* Wv    = (const float*)d_in[7];   // [1024,768]
  const float* bv    = (const float*)d_in[8];
  float* out = (float*)d_out;

  // Workspace layout (f16 unless noted):
  _Float16* qb  = (_Float16*)d_ws;                       // [32768][1024]
  _Float16* kb  = qb  + (size_t)32768 * 1024;            // [32768][1024]
  _Float16* vbt = kb  + (size_t)32768 * 1024;            // [16][1024][2048]
  _Float16* SP  = vbt + (size_t)32768 * 1024;            // [16][2048][2048]
  float* rowsum = (float*)(SP + (size_t)NBAT * LQ * LQ); // [32768] f32

  const dim3 blk(256);
  // Projections: grid (M/128, N/64) = (256, 16)
  proj_kernel<<<dim3(256, 16), blk, 0, stream>>>(query, Wq, bq, qb, 1024, 0);
  proj_kernel<<<dim3(256, 16), blk, 0, stream>>>(key,   Wk, bk, kb,  768, 0);
  proj_kernel<<<dim3(256, 16), blk, 0, stream>>>(value, Wv, bv, vbt, 768, 1);
  // Scores + softmax: grid (q-tiles, batches) = (16, 16)
  attn_softmax_kernel<<<dim3(16, 16), blk, 0, stream>>>(qb, kb, SP, rowsum);
  // P @ V: grid (m-tiles, o-tiles, batches) = (16, 8, 16)
  attn_pv_kernel<<<dim3(16, 8, 16), blk, 0, stream>>>(SP, vbt, rowsum, out);
}
